// mSEModule_31860067401817
// MI455X (gfx1250) — matrix operand order repaired and checked by reading, not compiled, not run on hardware
//
// mSE temporal-attention module for MI455X (gfx1250), wave32 + WMMA bf16.
//
// Pipeline (all intermediates in d_ws):
//  prep : w1/w3/wss2/wss4 -> bf16 (wss packed [o][tap*Cr+c]); x -> bf16 [n][hw][c]
//  S1   : bott = BN1(w1 @ x)            WMMA GEMM, LDS-staged B panel
//  S2   : cb   = depthwise3x3(bott); d_f/d_b diffs -> bf16 [n][hw][c]
//  S3   : avgpool 7->3 ; s2 = BN(conv3x3 @3x3) ; s4 = BN(conv3x3 @7x7)   (implicit-GEMM WMMA)
//  S4   : sum3 = (d+s2up+s4)/3 (in place, bf16) ; y = w3 @ sum3 -> [n][49][1024] bf16
//  S5   : G = y y^T / 32 (WMMA) ; amix = 0.5*(softmax(Gf)+softmax(Gb)) -> bf16 [n][49][64]
//  S6   : vflat = temporal depthwise conv of x -> bf16 [n][c][64]
//  S7   : out = x + gamma * (amix @ vflat)   (WMMA, K=64)

#include <hip/hip_runtime.h>
#include <hip/hip_bf16.h>
#include <cstdint>

typedef __attribute__((ext_vector_type(16))) __bf16 v16bf;
typedef __attribute__((ext_vector_type(8)))  float  v8f;

union ABf { v16bf v; uint4 q[2]; unsigned short s[16]; };

__device__ __forceinline__ unsigned short f2bf(float f) {
  unsigned u = __float_as_uint(f);
  unsigned r = u + 0x7FFFu + ((u >> 16) & 1u);
  return (unsigned short)(r >> 16);
}
__device__ __forceinline__ float bf2f(unsigned short h) {
  return __uint_as_float(((unsigned)h) << 16);
}
__device__ __forceinline__ v8f v8zero() {
  v8f z;
  #pragma unroll
  for (int i = 0; i < 8; ++i) z[i] = 0.0f;
  return z;
}
__device__ __forceinline__ ABf frag_zero() {
  ABf z; z.q[0] = make_uint4(0u,0u,0u,0u); z.q[1] = make_uint4(0u,0u,0u,0u);
  return z;
}
// A fragment (16x32 bf16): row ptr K-contiguous. lane: M=l&15, half=l>>4.
__device__ __forceinline__ ABf frag_a(const unsigned short* rowp, int k0, int half) {
  ABf a;
  a.q[0] = *reinterpret_cast<const uint4*>(rowp + k0 + 8 * half);
  a.q[1] = *reinterpret_cast<const uint4*>(rowp + k0 + 16 + 8 * half);
  return a;
}
// B fragment (32x16 bf16): column ptr K-contiguous. lane: N=l&15, half=l>>4.
__device__ __forceinline__ ABf frag_b(const unsigned short* colp, int k0, int half) {
  ABf b;
  const unsigned short* p = colp + k0 + 8 * half;
  b.q[0] = *reinterpret_cast<const uint4*>(p);
  b.q[1] = *reinterpret_cast<const uint4*>(p + 16);
  return b;
}
__device__ __forceinline__ v8f wmma_bf16(const ABf& a, const ABf& b, v8f c) {
  return __builtin_amdgcn_wmma_f32_16x16x32_bf16(false, a.v, false, b.v,
                                                 (short)0, c, false, false);
}

constexpr int NF = 512, C = 2048, CR = 1024, HH = 7, HW = 49, HWP = 9, QP = 64;
constexpr int KW = CR * 9;   // 9216
constexpr float EPS = 1e-5f;

// ---- ws layout (bytes) ----
constexpr size_t SZ_W1B  = (size_t)CR * C * 2;
constexpr size_t SZ_WSSB = (size_t)CR * CR * 9 * 2;
constexpr size_t SZ_W3B  = (size_t)CR * CR * 2;
constexpr size_t SZ_XB   = (size_t)NF * HW * C * 2;
constexpr size_t SZ_BUF  = (size_t)NF * CR * HW * 4;
constexpr size_t SZ_DB   = (size_t)NF * HW * CR * 2;
constexpr size_t SZ_PB   = (size_t)NF * HWP * CR * 2;
constexpr size_t SZ_S2   = (size_t)NF * CR * HWP * 4;
constexpr size_t SZ_LG   = (size_t)NF * HW * HW * 4;
constexpr size_t SZ_AM   = (size_t)NF * HW * QP * 2;
constexpr size_t SZ_VT   = (size_t)NF * C * QP * 2;

constexpr size_t OFF_W1B   = 0;
constexpr size_t OFF_WSS2B = OFF_W1B + SZ_W1B;
constexpr size_t OFF_WSS4B = OFF_WSS2B + SZ_WSSB;
constexpr size_t OFF_W3B   = OFF_WSS4B + SZ_WSSB;
constexpr size_t OFF_XB    = OFF_W3B + SZ_W3B;
constexpr size_t OFF_BUF1  = OFF_XB + SZ_XB;      // bott f32, later s4_f
constexpr size_t OFF_BUF2  = OFF_BUF1 + SZ_BUF;   // cb f32,   later s4_b
constexpr size_t OFF_DFB   = OFF_BUF2 + SZ_BUF;   // d_f bf16, later sum3_f
constexpr size_t OFF_DBB   = OFF_DFB + SZ_DB;
constexpr size_t OFF_PFB   = OFF_DBB + SZ_DB;
constexpr size_t OFF_PBB   = OFF_PFB + SZ_PB;
constexpr size_t OFF_S2F   = OFF_PBB + SZ_PB;
constexpr size_t OFF_S2B   = OFF_S2F + SZ_S2;
constexpr size_t OFF_LGF   = OFF_S2B + SZ_S2;
constexpr size_t OFF_LGB   = OFF_LGF + SZ_LG;
constexpr size_t OFF_AMIX  = OFF_LGB + SZ_LG;
constexpr size_t OFF_VTB   = OFF_AMIX + SZ_AM;
constexpr size_t OFF_YF    = OFF_XB;              // reuse xb region after S1
constexpr size_t OFF_YB    = OFF_XB + SZ_DB;

// ======================= prep kernels =======================
__global__ void k_cvt_plain(const float* __restrict__ src,
                            unsigned short* __restrict__ dst, int n) {
  int i = blockIdx.x * 256 + threadIdx.x;
  if (i < n) dst[i] = f2bf(src[i]);
}

// OIHW (o,c,3,3) -> [o][tap*CR + c]
__global__ void k_cvt_wss(const float* __restrict__ src,
                          unsigned short* __restrict__ dst) {
  int i = blockIdx.x * 256 + threadIdx.x;
  if (i >= CR * CR * 9) return;
  int t = i % 9, c = (i / 9) % CR, o = i / (9 * CR);
  dst[(size_t)o * KW + t * CR + c] = f2bf(src[i]);
}

// x (n,c,hw) f32 -> xb [n][hw][c] bf16
__global__ void k_cvt_x(const float* __restrict__ x,
                        unsigned short* __restrict__ xb) {
  int i = blockIdx.x * 256 + threadIdx.x;
  if (i >= NF * HW * C) return;
  int c = i & (C - 1), hw = (i >> 11) % HW, n = i / (C * HW);
  xb[i] = f2bf(x[((size_t)n * C + c) * HW + hw]);
}

// ======================= S1: bottleneck GEMM + BN1 =======================
// grid (4 hwTile, 8 oGroup, NF) x 256 threads (8 waves); LDS-staged B panel.
__global__ __launch_bounds__(256)
void k_gemm_bott(const unsigned short* __restrict__ xb,
                 const unsigned short* __restrict__ w1b,
                 const float* __restrict__ g, const float* __restrict__ bb,
                 const float* __restrict__ mm, const float* __restrict__ vv,
                 float* __restrict__ bott) {
  __shared__ unsigned short Bs[16 * 2048];  // 64KB panel: 16 px cols x K=2048
  const int n = blockIdx.z, hwT = blockIdx.x;
  // cooperative panel load (4096 uint4)
  #pragma unroll
  for (int it = 0; it < 16; ++it) {
    int lin = it * 256 + threadIdx.x;
    int col = lin >> 8, chunk = lin & 255;
    int gcol = hwT * 16 + col; if (gcol > 48) gcol = 48;
    reinterpret_cast<uint4*>(Bs)[col * 256 + chunk] =
        *reinterpret_cast<const uint4*>(xb + ((size_t)n * HW + gcol) * C + chunk * 8);
  }
  __syncthreads();
  const int wave = threadIdx.x >> 5, lane = threadIdx.x & 31;
  const int half = lane >> 4, lm = lane & 15;
  const int oT = blockIdx.y * 8 + wave, o0 = oT * 16;
  const unsigned short* arow = w1b + (size_t)(o0 + lm) * C;
  const unsigned short* bcol = Bs + lm * 2048;
  v8f acc = v8zero();
  #pragma unroll 4
  for (int kk = 0; kk < C / 32; ++kk) {
    ABf a = frag_a(arow, kk * 32, half);
    ABf b = frag_b(bcol, kk * 32, half);
    acc = wmma_bf16(a, b, acc);
  }
  const int col = hwT * 16 + lm;
  if (col < HW) {
    #pragma unroll
    for (int r = 0; r < 8; ++r) {
      int o = o0 + half * 8 + r;
      float sc = g[o] * rsqrtf(vv[o] + EPS);
      bott[((size_t)n * CR + o) * HW + col] = (acc[r] - mm[o]) * sc + bb[o];
    }
  }
}

// ======================= S2: depthwise 3x3 + temporal diffs =======================
__global__ void k_dwconv(const float* __restrict__ bott,
                         const float* __restrict__ w2, float* __restrict__ cb) {
  int i = blockIdx.x * 256 + threadIdx.x;
  if (i >= NF * CR * HW) return;
  int hw = i % HW, c = (i / HW) & (CR - 1), n = i / (HW * CR);
  int y = hw / HH, x = hw % HH;
  const float* src = bott + ((size_t)n * CR + c) * HW;
  float acc = 0.f;
  #pragma unroll
  for (int ky = 0; ky < 3; ++ky) {
    #pragma unroll
    for (int kx = 0; kx < 3; ++kx) {
      int sy = y + ky - 1, sx = x + kx - 1;
      if (sy >= 0 && sy < HH && sx >= 0 && sx < HH)
        acc += w2[c * 9 + ky * 3 + kx] * src[sy * HH + sx];
    }
  }
  cb[i] = acc;
}

__global__ void k_diff(const float* __restrict__ bott, const float* __restrict__ cb,
                       unsigned short* __restrict__ dfb, unsigned short* __restrict__ dbb) {
  int i = blockIdx.x * 256 + threadIdx.x;
  if (i >= NF * HW * CR) return;
  int c = i & (CR - 1), hw = (i >> 10) % HW, n = i / (CR * HW);
  int t = n & 7;
  size_t s = ((size_t)n * CR + c) * HW + hw;
  float bt = bott[s];
  float df = (t < 7) ? (cb[s + (size_t)CR * HW] - bt) : 0.f;
  float db = (t > 0) ? (cb[s - (size_t)CR * HW] - bt) : 0.f;
  dfb[i] = f2bf(df);  // [n][hw][c]
  dbb[i] = f2bf(db);
}

// ======================= S3: pool + the two 3x3 convs (implicit GEMM) ==========
__global__ void k_pool(const unsigned short* __restrict__ d,
                       unsigned short* __restrict__ p) {
  int i = blockIdx.x * 256 + threadIdx.x;
  if (i >= NF * HWP * CR) return;
  int c = i & (CR - 1), pp = (i >> 10) % HWP, n = i / (CR * HWP);
  int py = pp / 3, px = pp % 3;
  const unsigned short* src = d + (size_t)n * HW * CR;
  float s = 0.f;
  #pragma unroll
  for (int dy = 0; dy < 2; ++dy) {
    #pragma unroll
    for (int dx = 0; dx < 2; ++dx)
      s += bf2f(src[((2 * py + dy) * HH + (2 * px + dx)) * CR + c]);
  }
  p[i] = f2bf(0.25f * s);
}

// s2: conv3x3 on 3x3 pooled input, SAME pad; BN epilogue. grid (8, NF) x 256.
__global__ __launch_bounds__(256)
void k_conv_s2(const unsigned short* __restrict__ pin,
               const unsigned short* __restrict__ wb,
               const float* __restrict__ g, const float* __restrict__ bb,
               const float* __restrict__ mm, const float* __restrict__ vv,
               float* __restrict__ s2) {
  const int n = blockIdx.y;
  const int wave = threadIdx.x >> 5, lane = threadIdx.x & 31;
  const int half = lane >> 4, lm = lane & 15;
  const int oT = blockIdx.x * 8 + wave, o0 = oT * 16;
  const unsigned short* arow = wb + (size_t)(o0 + lm) * KW;
  const int pc = (lm < HWP) ? lm : HWP - 1;
  const int py = pc / 3, px = pc % 3;
  v8f acc = v8zero();
  for (int t = 0; t < 9; ++t) {
    int sy = py + t / 3 - 1, sx = px + t % 3 - 1;
    bool valid = (lm < HWP) && sy >= 0 && sy < 3 && sx >= 0 && sx < 3;
    const unsigned short* bcol = pin + ((size_t)n * HWP + (valid ? sy * 3 + sx : 0)) * CR;
    #pragma unroll 4
    for (int kc = 0; kc < CR / 32; ++kc) {
      ABf a = frag_a(arow + t * CR, kc * 32, half);
      ABf b = valid ? frag_b(bcol, kc * 32, half) : frag_zero();
      acc = wmma_bf16(a, b, acc);
    }
  }
  if (lm < HWP) {
    #pragma unroll
    for (int r = 0; r < 8; ++r) {
      int o = o0 + half * 8 + r;
      float sc = g[o] * rsqrtf(vv[o] + EPS);
      s2[((size_t)n * CR + o) * HWP + lm] = (acc[r] - mm[o]) * sc + bb[o];
    }
  }
}

// s4: conv3x3 on 7x7, SAME pad; BN epilogue. grid (4 colT, 8 oGrp, NF) x 256.
__global__ __launch_bounds__(256)
void k_conv_s4(const unsigned short* __restrict__ din,
               const unsigned short* __restrict__ wb,
               const float* __restrict__ g, const float* __restrict__ bb,
               const float* __restrict__ mm, const float* __restrict__ vv,
               float* __restrict__ s4) {
  const int n = blockIdx.z;
  const int wave = threadIdx.x >> 5, lane = threadIdx.x & 31;
  const int half = lane >> 4, lm = lane & 15;
  const int oT = blockIdx.y * 8 + wave, o0 = oT * 16;
  const unsigned short* arow = wb + (size_t)(o0 + lm) * KW;
  const int p = blockIdx.x * 16 + lm;
  const int pc = (p < HW) ? p : HW - 1;
  const int y = pc / HH, x = pc % HH;
  v8f acc = v8zero();
  for (int t = 0; t < 9; ++t) {
    int sy = y + t / 3 - 1, sx = x + t % 3 - 1;
    bool valid = (p < HW) && sy >= 0 && sy < HH && sx >= 0 && sx < HH;
    const unsigned short* bcol = din + ((size_t)n * HW + (valid ? sy * HH + sx : 0)) * CR;
    #pragma unroll 4
    for (int kc = 0; kc < CR / 32; ++kc) {
      ABf a = frag_a(arow + t * CR, kc * 32, half);
      ABf b = valid ? frag_b(bcol, kc * 32, half) : frag_zero();
      acc = wmma_bf16(a, b, acc);
    }
  }
  if (p < HW) {
    #pragma unroll
    for (int r = 0; r < 8; ++r) {
      int o = o0 + half * 8 + r;
      float sc = g[o] * rsqrtf(vv[o] + EPS);
      s4[((size_t)n * CR + o) * HW + p] = (acc[r] - mm[o]) * sc + bb[o];
    }
  }
}

// ======================= S4: sum3 (in place) + w3 GEMM =======================
__global__ void k_sum3(unsigned short* __restrict__ d,
                       const float* __restrict__ s2, const float* __restrict__ s4) {
  int i = blockIdx.x * 256 + threadIdx.x;
  if (i >= NF * HW * CR) return;
  int c = i & (CR - 1), hw = (i >> 10) % HW, n = i / (CR * HW);
  int y = hw / HH, x = hw % HH;
  int py = (y * 3) / HH, px = (x * 3) / HH;   // nearest upsample 3->7
  float dv = bf2f(d[i]);
  float s2v = s2[((size_t)n * CR + c) * HWP + py * 3 + px];
  float s4v = s4[((size_t)n * CR + c) * HW + hw];
  d[i] = f2bf((dv + s2v + s4v) * (1.0f / 3.0f));
}

// y[n][p][o] = sum_c w3[o][c] * sum3[n][p][c]. grid (4 pT, 8 oGrp, NF) x 256.
__global__ __launch_bounds__(256)
void k_gemm_w3(const unsigned short* __restrict__ sum3,
               const unsigned short* __restrict__ w3b,
               unsigned short* __restrict__ y) {
  const int n = blockIdx.z;
  const int wave = threadIdx.x >> 5, lane = threadIdx.x & 31;
  const int half = lane >> 4, lm = lane & 15;
  const int oT = blockIdx.y * 8 + wave, o0 = oT * 16;
  const unsigned short* arow = w3b + (size_t)(o0 + lm) * CR;
  const int p = blockIdx.x * 16 + lm;
  const unsigned short* bcol = sum3 + ((size_t)n * HW + ((p < HW) ? p : HW - 1)) * CR;
  v8f acc = v8zero();
  #pragma unroll 4
  for (int kc = 0; kc < CR / 32; ++kc) {
    ABf a = frag_a(arow, kc * 32, half);
    ABf b = frag_b(bcol, kc * 32, half);
    acc = wmma_bf16(a, b, acc);
  }
  if (p < HW) {
    #pragma unroll
    for (int r = 0; r < 8; ++r) {
      int o = o0 + half * 8 + r;
      y[((size_t)n * HW + p) * CR + o] = f2bf(acc[r]);
    }
  }
}

// ======================= S5: Gram + softmax-mix =======================
// logits[n][p][q] = (y_p . y_q) / 32. grid (2, NF) x 256; tile = bx*8+wave.
__global__ __launch_bounds__(256)
void k_gram(const unsigned short* __restrict__ y, float* __restrict__ lg) {
  const int n = blockIdx.y;
  const int wave = threadIdx.x >> 5, lane = threadIdx.x & 31;
  const int half = lane >> 4, lm = lane & 15;
  const int tile = blockIdx.x * 8 + wave;
  const int pT = tile >> 2, qT = tile & 3;
  const int pr = pT * 16 + lm, qc = qT * 16 + lm;
  const unsigned short* arow = y + ((size_t)n * HW + ((pr < HW) ? pr : HW - 1)) * CR;
  const unsigned short* bcol = y + ((size_t)n * HW + ((qc < HW) ? qc : HW - 1)) * CR;
  v8f acc = v8zero();
  #pragma unroll 4
  for (int kc = 0; kc < CR / 32; ++kc) {
    ABf a = frag_a(arow, kc * 32, half);
    ABf b = frag_b(bcol, kc * 32, half);
    acc = wmma_bf16(a, b, acc);
  }
  #pragma unroll
  for (int r = 0; r < 8; ++r) {
    int p = pT * 16 + half * 8 + r;
    if (p < HW && qc < HW)
      lg[((size_t)n * HW + p) * HW + qc] = acc[r] * 0.03125f;  // 1/sqrt(1024)
  }
}

__global__ void k_softmax_mix(const float* __restrict__ lf, const float* __restrict__ lb,
                              unsigned short* __restrict__ amix) {
  int i = blockIdx.x * 256 + threadIdx.x;
  if (i >= NF * HW) return;
  int n = i / HW, p = i % HW;
  const float* rf = lf + ((size_t)n * HW + p) * HW;
  const float* rb = lb + ((size_t)n * HW + p) * HW;
  float mf = -1e30f, mb = -1e30f;
  for (int q = 0; q < HW; ++q) { mf = fmaxf(mf, rf[q]); mb = fmaxf(mb, rb[q]); }
  float sf = 0.f, sb = 0.f;
  for (int q = 0; q < HW; ++q) { sf += __expf(rf[q] - mf); sb += __expf(rb[q] - mb); }
  float inf_ = 1.f / sf, inb_ = 1.f / sb;
  unsigned short* ar = amix + ((size_t)n * HW + p) * QP;
  for (int q = 0; q < HW; ++q)
    ar[q] = f2bf(0.5f * (__expf(rf[q] - mf) * inf_ + __expf(rb[q] - mb) * inb_));
  for (int q = HW; q < QP; ++q) ar[q] = 0;
}

// ======================= S6: temporal depthwise conv -> vflat bf16 =============
__global__ void k_temporal(const float* __restrict__ x, const float* __restrict__ wt,
                           unsigned short* __restrict__ vtb) {
  int i = blockIdx.x * 256 + threadIdx.x;
  if (i >= NF * C * QP) return;
  int q = i & 63, c = (i >> 6) & (C - 1), n = i >> 17;
  float val = 0.f;
  if (q < HW) {
    int t = n & 7;
    size_t base = ((size_t)n * C + c) * HW + q;
    val = wt[c * 3 + 1] * x[base];
    if (t > 0) val += wt[c * 3 + 0] * x[base - (size_t)C * HW];
    if (t < 7) val += wt[c * 3 + 2] * x[base + (size_t)C * HW];
  }
  vtb[i] = f2bf(val);  // [n][c][64], q>=49 zero-padded
}

// ======================= S7: attention GEMM + residual =======================
// act[n][p][c] = sum_q amix[n][p][q] * vflat[n][q][c]; out = x + gamma*act (NCHW).
// grid (16 cGrp, 4 pT, NF) x 256; wave -> cTile.
__global__ __launch_bounds__(256)
void k_attn_out(const unsigned short* __restrict__ amix,
                const unsigned short* __restrict__ vtb,
                const float* __restrict__ x, const float* __restrict__ gamma,
                float* __restrict__ out) {
  const int n = blockIdx.z, pT = blockIdx.y;
  const int wave = threadIdx.x >> 5, lane = threadIdx.x & 31;
  const int half = lane >> 4, lm = lane & 15;
  const int cT = blockIdx.x * 8 + wave;
  const int p = pT * 16 + lm;
  const unsigned short* arow = amix + ((size_t)n * HW + ((p < HW) ? p : HW - 1)) * QP;
  const int cch = cT * 16 + lm;
  const unsigned short* bcol = vtb + ((size_t)n * C + cch) * QP;
  v8f acc = v8zero();
  #pragma unroll
  for (int kk = 0; kk < QP / 32; ++kk) {
    ABf a = frag_a(arow, kk * 32, half);
    ABf b = frag_b(bcol, kk * 32, half);
    acc = wmma_bf16(a, b, acc);
  }
  const float gm = *gamma;
  #pragma unroll
  for (int r = 0; r < 8; ++r) {
    int pr = pT * 16 + half * 8 + r;
    if (pr < HW) {
      size_t idx = ((size_t)n * C + cch) * HW + pr;
      out[idx] = x[idx] + gm * acc[r];
    }
  }
}

// ======================= host launcher =======================
extern "C" void kernel_launch(void* const* d_in, const int* in_sizes, int n_in,
                              void* d_out, int out_size, void* d_ws, size_t ws_size,
                              hipStream_t stream) {
  const float* x     = (const float*)d_in[0];
  const float* w1    = (const float*)d_in[1];
  const float* bn1g  = (const float*)d_in[2];
  const float* bn1b  = (const float*)d_in[3];
  const float* bn1m  = (const float*)d_in[4];
  const float* bn1v  = (const float*)d_in[5];
  const float* w2    = (const float*)d_in[6];
  const float* wss2  = (const float*)d_in[7];
  const float* b2g   = (const float*)d_in[8];
  const float* b2b   = (const float*)d_in[9];
  const float* b2m   = (const float*)d_in[10];
  const float* b2v   = (const float*)d_in[11];
  const float* wss4  = (const float*)d_in[12];
  const float* b4g   = (const float*)d_in[13];
  const float* b4b   = (const float*)d_in[14];
  const float* b4m   = (const float*)d_in[15];
  const float* b4v   = (const float*)d_in[16];
  const float* w3    = (const float*)d_in[17];
  const float* wt    = (const float*)d_in[18];
  const float* gamma = (const float*)d_in[19];
  float* out = (float*)d_out;

  char* ws = (char*)d_ws;
  unsigned short* w1b   = (unsigned short*)(ws + OFF_W1B);
  unsigned short* wss2b = (unsigned short*)(ws + OFF_WSS2B);
  unsigned short* wss4b = (unsigned short*)(ws + OFF_WSS4B);
  unsigned short* w3b   = (unsigned short*)(ws + OFF_W3B);
  unsigned short* xb    = (unsigned short*)(ws + OFF_XB);
  float*          buf1  = (float*)(ws + OFF_BUF1);
  float*          buf2  = (float*)(ws + OFF_BUF2);
  unsigned short* dfb   = (unsigned short*)(ws + OFF_DFB);
  unsigned short* dbb   = (unsigned short*)(ws + OFF_DBB);
  unsigned short* pfb   = (unsigned short*)(ws + OFF_PFB);
  unsigned short* pbb   = (unsigned short*)(ws + OFF_PBB);
  float*          s2f   = (float*)(ws + OFF_S2F);
  float*          s2b   = (float*)(ws + OFF_S2B);
  float*          lgf   = (float*)(ws + OFF_LGF);
  float*          lgb   = (float*)(ws + OFF_LGB);
  unsigned short* amix  = (unsigned short*)(ws + OFF_AMIX);
  unsigned short* vtb   = (unsigned short*)(ws + OFF_VTB);
  unsigned short* yf    = (unsigned short*)(ws + OFF_YF);
  unsigned short* yb    = (unsigned short*)(ws + OFF_YB);

  auto blk = [](long long n) { return (unsigned)((n + 255) / 256); };

  // prep
  k_cvt_plain<<<blk((long long)CR * C), 256, 0, stream>>>(w1, w1b, CR * C);
  k_cvt_plain<<<blk((long long)CR * CR), 256, 0, stream>>>(w3, w3b, CR * CR);
  k_cvt_wss<<<blk((long long)CR * CR * 9), 256, 0, stream>>>(wss2, wss2b);
  k_cvt_wss<<<blk((long long)CR * CR * 9), 256, 0, stream>>>(wss4, wss4b);
  k_cvt_x<<<blk((long long)NF * HW * C), 256, 0, stream>>>(x, xb);
  // S1
  k_gemm_bott<<<dim3(4, 8, NF), 256, 0, stream>>>(xb, w1b, bn1g, bn1b, bn1m, bn1v, buf1);
  // S2
  k_dwconv<<<blk((long long)NF * CR * HW), 256, 0, stream>>>(buf1, w2, buf2);
  k_diff<<<blk((long long)NF * HW * CR), 256, 0, stream>>>(buf1, buf2, dfb, dbb);
  // S3
  k_pool<<<blk((long long)NF * HWP * CR), 256, 0, stream>>>(dfb, pfb);
  k_pool<<<blk((long long)NF * HWP * CR), 256, 0, stream>>>(dbb, pbb);
  k_conv_s2<<<dim3(8, NF), 256, 0, stream>>>(pfb, wss2b, b2g, b2b, b2m, b2v, s2f);
  k_conv_s2<<<dim3(8, NF), 256, 0, stream>>>(pbb, wss2b, b2g, b2b, b2m, b2v, s2b);
  k_conv_s4<<<dim3(4, 8, NF), 256, 0, stream>>>(dfb, wss4b, b4g, b4b, b4m, b4v, buf1);
  k_conv_s4<<<dim3(4, 8, NF), 256, 0, stream>>>(dbb, wss4b, b4g, b4b, b4m, b4v, buf2);
  // S4
  k_sum3<<<blk((long long)NF * HW * CR), 256, 0, stream>>>(dfb, s2f, buf1);
  k_sum3<<<blk((long long)NF * HW * CR), 256, 0, stream>>>(dbb, s2b, buf2);
  k_gemm_w3<<<dim3(4, 8, NF), 256, 0, stream>>>(dfb, w3b, yf);
  k_gemm_w3<<<dim3(4, 8, NF), 256, 0, stream>>>(dbb, w3b, yb);
  // S5
  k_gram<<<dim3(2, NF), 256, 0, stream>>>(yf, lgf);
  k_gram<<<dim3(2, NF), 256, 0, stream>>>(yb, lgb);
  k_softmax_mix<<<blk((long long)NF * HW), 256, 0, stream>>>(lgf, lgb, amix);
  // S6
  k_temporal<<<blk((long long)NF * C * QP), 256, 0, stream>>>(x, wt, vtb);
  // S7
  k_attn_out<<<dim3(16, 4, NF), 256, 0, stream>>>(amix, vtb, x, gamma, out);
  (void)in_sizes; (void)n_in; (void)out_size; (void)ws_size;
}
